// ReadoutLayer_10617159156210
// MI455X (gfx1250) — matrix operand order repaired and checked by reading, not compiled, not run on hardware
//
#include <hip/hip_runtime.h>
#include <hip/hip_bf16.h>

// ---------------------------------------------------------------------------
// Fused ReadoutLayer for MI455X (gfx1250, wave32, WMMA bf16 16x16x32)
//   Wx = x @ W^T   (streamed, never hits HBM)
//   ut = a*ut + (1-a)*wx ; out += (t>=100) * softmax(ut)
// ---------------------------------------------------------------------------

typedef __attribute__((ext_vector_type(16))) __bf16 v16bf;
typedef __attribute__((ext_vector_type(8)))  float  v8f;

#define B_      128
#define T_      1000
#define D_      1024
#define H_      256
#define TCROP   100
#define KSTEPS  (D_ / 32)        // 32 K-steps of 32 for bf16 WMMA
#define NTILES  (H_ / 16)        // 16 N-tiles
#define TTILES  ((T_ + 15) / 16) // 63 time tiles

#define ALPHA_MIN_C 0.8187307530779818f   // exp(-1/5)
#define ALPHA_MAX_C 0.9607894391523232f   // exp(-1/25)

// fp32 -> bf16 round-to-nearest-even via bit manipulation (no libcall)
__device__ __forceinline__ __bf16 f2bf(float f) {
    unsigned u = __builtin_bit_cast(unsigned, f);
    unsigned r = u + 0x7FFFu + ((u >> 16) & 1u);
    unsigned short s = (unsigned short)(r >> 16);
    return __builtin_bit_cast(__bf16, s);
}

// ---------------------------------------------------------------------------
// Pack W [H=256, D=1024] fp32 into bf16 B-fragment-major layout:
//   Wb[nt][kk][lane][i],  nt=0..15, kk=0..31, lane=0..31, i=0..15
//   B element (k, n): n = nt*16 + (lane&15), k = kk*32 + 16*(lane>>4) + i
// Each WMMA lane then loads its B fragment as 32 contiguous bytes.
// Total 512 KB, lives in workspace (L2-resident on MI455X).
// ---------------------------------------------------------------------------
__global__ void pack_w_bf16(const float* __restrict__ W, __bf16* __restrict__ Wb) {
    int e    = blockIdx.x * blockDim.x + threadIdx.x;   // 0 .. 262143
    int i    = e & 15;
    int lane = (e >> 4) & 31;
    int kk   = (e >> 9) & 31;
    int nt   = e >> 14;
    int half = lane >> 4;
    int n    = nt * 16 + (lane & 15);
    int k    = kk * 32 + 16 * half + i;
    Wb[e] = f2bf(W[n * D_ + k]);
}

// ---------------------------------------------------------------------------
// Main fused kernel: one workgroup (8 wave32) per batch element.
// ---------------------------------------------------------------------------
__global__ __launch_bounds__(256)
void readout_fused(const float* __restrict__ x, const __bf16* __restrict__ Wb,
                   const float* __restrict__ alpha, float* __restrict__ out) {
    // A fragments for a 16(t) x 1024(k) tile, fragment-major: [kk][lane][16]
    __shared__ __align__(32) __bf16 xs[KSTEPS * 32 * 16];   // 32 KB
    __shared__ float wxs[16 * H_];                          // 16 KB (16 t x 256 h)
    __shared__ float redm[8];
    __shared__ float reds[8];

    const int tid  = threadIdx.x;
    const int lane = tid & 31;
    const int wv   = tid >> 5;        // wave 0..7
    const int half = lane >> 4;
    const int b    = blockIdx.x;
    const int h    = tid;             // thread <-> feature channel (H_ == blockDim)

    float a = alpha[h];
    a = fminf(fmaxf(a, ALPHA_MIN_C), ALPHA_MAX_C);
    const float one_m_a = 1.0f - a;
    float ut  = 0.0f;
    float acc = 0.0f;

    const float* xb = x + (size_t)b * T_ * D_;

    for (int tt = 0; tt < TTILES; ++tt) {
        const int t0 = tt * 16;

        // ---- stage x tile fp32 -> bf16 into A-fragment order -------------
        // A layout per lane as staged here (same k order as Wb pack): row M = lane&15, half = lane>>4
        //   element i -> k_local = 16*half + i
        for (int e = tid; e < KSTEPS * 32 * 16; e += 256) {
            int i   = e & 15;
            int ln  = (e >> 4) & 31;
            int kk  = e >> 9;
            int r   = ln & 15;
            int hf  = ln >> 4;
            int klc = 16 * hf + i;
            int k   = kk * 32 + klc;
            int t   = t0 + r;
            float v = (t < T_) ? xb[(size_t)t * D_ + k] : 0.0f;
            xs[e] = f2bf(v);
            // prefetch next time-tile of x (gfx1250 global_prefetch_b8)
            if ((i == 0) && (t + 16 < T_))
                __builtin_prefetch(&xb[(size_t)(t + 16) * D_ + k], 0, 1);
        }
        __syncthreads();

        // ---- WMMA: each wave computes two 16x16 C tiles (its H slice) ----
        for (int s = 0; s < 2; ++s) {
            const int nt = wv * 2 + s;
            v8f c = {};
            const v16bf* ap = (const v16bf*)(xs) + lane;
            const v16bf* bp = (const v16bf*)(Wb) + (size_t)nt * KSTEPS * 32 + lane;
            #pragma unroll 8
            for (int kk = 0; kk < KSTEPS; ++kk) {
                v16bf af = ap[kk * 32];   // 32B contiguous LDS read
                v16bf bf = bp[kk * 32];   // 32B contiguous global (L2) read
                c = __builtin_amdgcn_wmma_f32_16x16x32_bf16(
                        /*neg_a=*/false, af, /*neg_b=*/false, bf,
                        /*c_mod=*/(short)0, c, /*reuse_a=*/false, /*reuse_b=*/false);
            }
            // C layout: VGPR j -> M = j + 8*half, N = lane&15
            const int hn = nt * 16 + (lane & 15);
            #pragma unroll
            for (int j = 0; j < 8; ++j)
                wxs[(j + 8 * half) * H_ + hn] = c[j];
        }
        __syncthreads();

        // ---- sequential EMA + softmax scan over the 16 timesteps ---------
        const int tmax = min(16, T_ - t0);
        for (int tl = 0; tl < tmax; ++tl) {
            float wx = wxs[tl * H_ + h];
            ut = a * ut + one_m_a * wx;

            // block max over H=256: wave32 shfl tree + 8 LDS partials
            float v = ut;
            #pragma unroll
            for (int o = 16; o > 0; o >>= 1) v = fmaxf(v, __shfl_xor(v, o, 32));
            if (lane == 0) redm[wv] = v;
            __syncthreads();
            float m = redm[0];
            #pragma unroll
            for (int q = 1; q < 8; ++q) m = fmaxf(m, redm[q]);

            float p = __expf(ut - m);

            float sv = p;
            #pragma unroll
            for (int o = 16; o > 0; o >>= 1) sv += __shfl_xor(sv, o, 32);
            if (lane == 0) reds[wv] = sv;
            __syncthreads();
            float ssum = 0.0f;
            #pragma unroll
            for (int q = 0; q < 8; ++q) ssum += reds[q];

            if (t0 + tl >= TCROP) acc += p / ssum;
        }
        __syncthreads();  // before next tile overwrites xs/wxs
    }

    out[b * H_ + h] = acc;
}

// ---------------------------------------------------------------------------
extern "C" void kernel_launch(void* const* d_in, const int* in_sizes, int n_in,
                              void* d_out, int out_size, void* d_ws, size_t ws_size,
                              hipStream_t stream) {
    const float* x     = (const float*)d_in[0];   // [128, 1000, 1024]
    const float* W     = (const float*)d_in[1];   // [256, 1024]
    const float* alpha = (const float*)d_in[2];   // [256]
    float* out = (float*)d_out;                   // [128, 256]

    __bf16* Wb = (__bf16*)d_ws;                   // 512 KB packed bf16 W^T

    // repack W every call (deterministic, cheap, graph-capture safe)
    pack_w_bf16<<<(NTILES * KSTEPS * 32 * 16) / 256, 256, 0, stream>>>(W, Wb);
    readout_fused<<<B_, 256, 0, stream>>>(x, Wb, alpha, out);
}
